// CondConv2d_3650722202213
// MI455X (gfx1250) — matrix operand order repaired and checked by reading, not compiled, not run on hardware
//
#include <hip/hip_runtime.h>
#include <hip/hip_bf16.h>

typedef __attribute__((ext_vector_type(16))) __bf16 v16bf;
typedef __attribute__((ext_vector_type(8)))  float  v8f;

#define BSZ    32
#define ESZ    8
#define CIN_C  256
#define COUT_C 256
#define HIMG   56
#define WIMG   56
#define NPAR   (COUT_C * CIN_C * 9)   // 589824
#define CICH   64                     // ci chunk staged in LDS

__device__ __forceinline__ unsigned short f2bf(float f) {
    unsigned u = __builtin_bit_cast(unsigned, f);
    u += 0x7FFFu + ((u >> 16) & 1u);   // round-to-nearest-even
    return (unsigned short)(u >> 16);
}

// ---------------------------------------------------------------------------
// Kernel 1: per-sample expert mixing, f32 -> bf16, transposed to [co][tap][ci]
// so the conv kernel's A-fragments (16 co x 32 contiguous ci for one tap) are
// contiguous 16-byte loads.
// ---------------------------------------------------------------------------
__global__ __launch_bounds__(256) void mix_weights(
    const float* __restrict__ routing,    // [B,E]
    const float* __restrict__ experts,    // [E, NPAR]  (co,ci,kh,kw)
    unsigned short* __restrict__ wmix)    // [B, co*9*256 + t*256 + ci] bf16
{
    const int b = blockIdx.y;
    const long long idx = (long long)blockIdx.x * blockDim.x + threadIdx.x;
    if (idx >= NPAR) return;
    const int ci = (int)(idx % CIN_C);
    const int t  = (int)((idx / CIN_C) % 9);
    const int co = (int)(idx / (CIN_C * 9));
    const long long src = ((long long)co * CIN_C + ci) * 9 + t;

    float acc = 0.f;
#pragma unroll
    for (int e = 0; e < ESZ; ++e)
        acc += routing[b * ESZ + e] * experts[(long long)e * NPAR + src];

    wmix[(long long)b * NPAR + idx] = f2bf(acc);
}

// ---------------------------------------------------------------------------
// Kernel 2: implicit-GEMM CondConv.
// Block = 256 threads (8 wave32) handles (b, 2 output rows) with ALL 256 co.
// GEMM per block: M=256 (co), N=112 (2*56 pixels), K=2304 (ci * 9 taps).
// Wave w owns co rows [32w, 32w+32): TWO 16-row A fragments, so every LDS
// B fragment feeds two v_wmma -> ds:wmma ratio 1:1.
// Accumulators: 2 (m) x 7 (n-tiles) x v8f = 112 VGPRs/lane.
// ---------------------------------------------------------------------------
__global__ __launch_bounds__(256) void condconv_wmma(
    const float* __restrict__ x,          // [B, CIN, 56, 56]
    const unsigned short* __restrict__ wmix,
    float* __restrict__ out)              // [B, COUT, 56, 56]
{
    // LDS x tile: [hl 0..3][wl 0..57][ci 0..63] bf16 (ci innermost => B frags contiguous)
    __shared__ unsigned short lds[4 * 58 * CICH];   // 29,696 B

    const int bid = blockIdx.x;
    const int b   = bid / 28;             // 28 two-row strips per sample
    const int h0  = (bid % 28) * 2;

    const int tid    = threadIdx.x;
    const int wave   = tid >> 5;
    const int lane   = tid & 31;
    const int lanelo = lane & 15;
    const int hihalf = lane >> 4;

    // per-lane N-tile coordinates (n = j*16 + lanelo within the 112-pixel strip)
    int rj[7], wj[7];
#pragma unroll
    for (int j = 0; j < 7; ++j) { int n = j * 16 + lanelo; rj[j] = n / 56; wj[j] = n % 56; }

    v8f acc[2][7];
#pragma unroll
    for (int m = 0; m < 2; ++m)
#pragma unroll
        for (int j = 0; j < 7; ++j) acc[m][j] = (v8f){0.f,0.f,0.f,0.f,0.f,0.f,0.f,0.f};

    const int arow0 = wave * 32 + lanelo;                 // A rows for m=0 fragment
    const unsigned short* wbase = wmix + (long long)b * NPAR;
    const int koffA = hihalf ? 8  : 0;   // 16-bit A 16x32 lane layout
    const int koffB = hihalf ? 8  : 0;   // 16-bit B 32x16 lane layout

    const long long xbase = (long long)b * CIN_C * (HIMG * WIMG);

    for (int cc = 0; cc < 4; ++cc) {                     // ci chunks of 64
        __syncthreads();
        // stage halo tile: rows h0-1..h0+2, cols -1..56, ci chunk; zero-pad OOB
        for (int i = tid; i < 232 * CICH; i += 256) {
            const int ci_l = i / 232;
            const int pos  = i % 232;
            const int hl = pos / 58, wl = pos % 58;
            const int h = h0 + hl - 1, w = wl - 1;
            float v = 0.f;
            if (h >= 0 && h < HIMG && w >= 0 && w < WIMG)
                v = x[xbase + ((long long)(cc * CICH + ci_l) * HIMG + h) * WIMG + w];
            lds[(hl * 58 + wl) * CICH + ci_l] = f2bf(v);
        }
        // prefetch next ci chunk of x into cache (gfx1250 global_prefetch_b8)
        if (cc < 3) {
            const float* pf = x + xbase
                + (long long)((cc + 1) * CICH + (tid >> 3)) * (HIMG * WIMG)
                + (long long)(h0 > 0 ? h0 - 1 : 0) * WIMG + (tid & 7) * 8;
            __builtin_prefetch(pf, 0, 1);
        }
        __syncthreads();

#pragma unroll
        for (int t = 0; t < 9; ++t) {                    // 3x3 taps
            const int kh = t / 3, kw = t % 3;
#pragma unroll
            for (int kk = 0; kk < 2; ++kk) {             // two K=32 steps per chunk
                const int ci0 = cc * CICH + kk * 32;

                // ---- A fragments: weights [16 co x 32 ci] for tap t, two M-tiles ----
                union { uint4 q[2]; v16bf v; } afrag[2];
#pragma unroll
                for (int m = 0; m < 2; ++m) {
                    const unsigned short* ap =
                        wbase + ((long long)((arow0 + m * 16) * 9 + t)) * CIN_C + ci0;
                    afrag[m].q[0] = *(const uint4*)(ap + koffA);        // K {0..7 | 8..15}
                    afrag[m].q[1] = *(const uint4*)(ap + koffA + 16);   // K {16..23 | 24..31}
                }

#pragma unroll
                for (int j = 0; j < 7; ++j) {
                    // ---- B fragment: im2col [32 ci x 16 px] from LDS ----
                    const int hl = rj[j] + kh;
                    const int wl = wj[j] + kw;
                    const unsigned short* bp =
                        &lds[(hl * 58 + wl) * CICH + kk * 32 + koffB];
                    union { uint4 q[2]; v16bf v; } bfrag;
                    bfrag.q[0] = *(const uint4*)(bp);            // K +0..7
                    bfrag.q[1] = *(const uint4*)(bp + 16);       // K +16..23

                    // one LDS B fragment feeds two WMMAs (ds:wmma = 1:1)
                    acc[0][j] = __builtin_amdgcn_wmma_f32_16x16x32_bf16(
                        false, afrag[0].v, false, bfrag.v,
                        (short)0, acc[0][j], false, false);
                    acc[1][j] = __builtin_amdgcn_wmma_f32_16x16x32_bf16(
                        false, afrag[1].v, false, bfrag.v,
                        (short)0, acc[1][j], false, false);
                }
            }
        }
    }

    // ---- store: C/D layout -> VGPR v holds row M=v (lanes 0-15) / v+8 (16-31)
#pragma unroll
    for (int m = 0; m < 2; ++m) {
#pragma unroll
        for (int j = 0; j < 7; ++j) {
            const int h = h0 + rj[j];
            const int w = wj[j];
#pragma unroll
            for (int v = 0; v < 8; ++v) {
                const int row = wave * 32 + m * 16 + v + hihalf * 8;
                out[(((long long)b * COUT_C + row) * HIMG + h) * WIMG + w] = acc[m][j][v];
            }
        }
    }
}

// ---------------------------------------------------------------------------
extern "C" void kernel_launch(void* const* d_in, const int* in_sizes, int n_in,
                              void* d_out, int out_size, void* d_ws, size_t ws_size,
                              hipStream_t stream) {
    const float* x        = (const float*)d_in[0];
    const float* routing  = (const float*)d_in[1];
    const float* experts  = (const float*)d_in[2];
    float* out            = (float*)d_out;
    unsigned short* wmix  = (unsigned short*)d_ws;   // 32*589824*2 = 37.75 MB

    dim3 g1(NPAR / 256, BSZ);                        // 2304 x 32 blocks
    mix_weights<<<g1, 256, 0, stream>>>(routing, experts, wmix);

    condconv_wmma<<<BSZ * 28, 256, 0, stream>>>(x, wmix, out);
}
